// myRNNModel_64476049047841
// MI455X (gfx1250) — matrix operand order repaired and checked by reading, not compiled, not run on hardware
//
#include <hip/hip_runtime.h>

// ---------------------------------------------------------------------------
// Types for CDNA5 WMMA (gfx1250, wave32)
// ---------------------------------------------------------------------------
typedef __attribute__((ext_vector_type(16))) __bf16 v16bf;
typedef __attribute__((ext_vector_type(8)))  float  v8f;

union V16 {
    v16bf v;
    uint4  q[2];   // two 16-byte halves
};

union F8 {
    float f[8];
    uint4 q[2];    // two 16-byte halves
};

__device__ __forceinline__ v8f v8f_zero() {
    v8f z;
#pragma unroll
    for (int i = 0; i < 8; ++i) z[i] = 0.0f;
    return z;
}

__device__ __forceinline__ v8f wmma_bf16(const V16& a, const V16& b, v8f c) {
    // D = A(16x32 bf16) * B(32x16 bf16) + C(16x16 f32)
    return __builtin_amdgcn_wmma_f32_16x16x32_bf16(
        /*neg_a=*/false, a.v, /*neg_b=*/false, b.v,
        /*c_mod=*/(short)0, c, /*reuse_a=*/false, /*reuse_b=*/false);
}

// Fast tanh via native exp2 pipeline: sign(x) * (1 - 2/(e^{2|x|}+1)).
// Saturates correctly (exp -> inf => result -> +-1); rounded to bf16 anyway.
__device__ __forceinline__ float fast_tanh(float x) {
    const float ax = fabsf(x);
    const float e  = __expf(2.0f * ax);
    const float r  = 1.0f - 2.0f / (e + 1.0f);
    return copysignf(r, x);
}

// ---------------------------------------------------------------------------
// Prep kernels: f32 -> bf16 convert (and transpose for B-side operands)
// ---------------------------------------------------------------------------
__global__ void k_convert_bf16(const float* __restrict__ in,
                               __bf16* __restrict__ out, int n) {
    int i = blockIdx.x * blockDim.x + threadIdx.x;
    if (i < n) out[i] = (__bf16)in[i];
}

// in: [K][N] row-major f32  ->  out: [N][K] row-major bf16 (K contiguous)
__global__ void k_transpose_bf16(const float* __restrict__ in,
                                 __bf16* __restrict__ out, int K, int N) {
    int i = blockIdx.x * blockDim.x + threadIdx.x;
    if (i >= K * N) return;
    int n = i / K;
    int k = i - n * K;
    out[i] = (__bf16)in[(size_t)k * N + n];
}

// ---------------------------------------------------------------------------
// Tiled bf16 WMMA GEMM:  C[M x N] = A[M x K] * BT[N x K]^T + bias
//   block = 256 threads = 8 waves; block tile = 128 x (NT*16)
//   B block staged into LDS with CDNA5 async global->LDS copies
//   A chunks double-buffered (global loads overlap WMMA), K fully unrolled
//   GATHER: A rows indirect through ids[] (embedding lookup)
//   XPMODE: output stored in the scan kernel's register layout:
//           xp_s[((t*2 + mt)*2 + half)][n][r]   (N==256, T==512, B==32)
//           where row = b*512+t, b = mt*16 + half*8 + r
// ---------------------------------------------------------------------------
template <int NT, int KDIM, bool GATHER, bool XPMODE>
__global__ __launch_bounds__(256) void k_gemm(
    const __bf16* __restrict__ A, const int* __restrict__ ids,
    const __bf16* __restrict__ BT, const float* __restrict__ bias,
    float* __restrict__ out, int N)
{
    constexpr int KC = KDIM / 32;
    extern __shared__ char smem[];
    const int tid   = threadIdx.x;
    const int lane  = tid & 31;
    const int wave  = tid >> 5;
    const int nbase = blockIdx.y * (NT * 16);
    const int mbase = blockIdx.x * 128 + wave * 16;

    // --- stage B block [NT*16][KDIM] bf16 into LDS (CDNA5 async copies) ---
    {
        const char* src = (const char*)(BT + (size_t)nbase * KDIM);
        constexpr int chunks = (NT * 16 * KDIM * 2) >> 4;   // 16B chunks
        for (int c = tid; c < chunks; c += 256) {
            unsigned long long ga = (unsigned long long)(src + (size_t)c * 16);
            unsigned int       la = (unsigned int)(size_t)&smem[c * 16];
            asm volatile("global_load_async_to_lds_b128 %0, %1, off"
                         :: "v"(la), "v"(ga) : "memory");
        }
        asm volatile("s_wait_asynccnt 0" ::: "memory");
    }
    __syncthreads();

    // A operand addressing (16-bit A 16x32 layout: lanes<16 hold K 0..7/16..23,
    // lanes>=16 hold K 8..15/24..31 of rows m = lane%16)
    const int mlane = lane & 15;
    const int mrow  = mbase + mlane;
    const size_t arow = (GATHER ? (size_t)ids[mrow] : (size_t)mrow) * (size_t)KDIM;
    const int ahalf = (lane >> 4) * 8;    // A K-offset per lane half
    const int bhalf = (lane >> 4) * 8;    // B K-offset per lane half

    auto loadA = [&](int kc, V16& a) {
        const __bf16* pa = A + arow + kc * 32 + ahalf;
        a.q[0] = *(const uint4*)pa;
        a.q[1] = *(const uint4*)(pa + 16);
    };

    v8f acc[NT];
#pragma unroll
    for (int nt = 0; nt < NT; ++nt) acc[nt] = v8f_zero();

    V16 a[2];
    loadA(0, a[0]);
#pragma unroll
    for (int kc = 0; kc < KC; ++kc) {
        const int pb = kc & 1;
        if (kc + 1 < KC) loadA(kc + 1, a[pb ^ 1]);   // prefetch next A chunk
        V16 b[NT];
#pragma unroll
        for (int nt = 0; nt < NT; ++nt) {
            const char* pbm = smem +
                (size_t)(((nt * 16 + mlane) * KDIM + kc * 32 + bhalf) * 2);
            b[nt].q[0] = *(const uint4*)pbm;
            b[nt].q[1] = *(const uint4*)(pbm + 32);
        }
#pragma unroll
        for (int nt = 0; nt < NT; ++nt)
            acc[nt] = wmma_bf16(a[pb], b[nt], acc[nt]);
    }

    // --- epilogue: bias add + store (C/D layout: VGPR r -> row r (+8 hi lanes)) ---
    const int moff = (lane >> 4) * 8;
#pragma unroll
    for (int nt = 0; nt < NT; ++nt) {
        const int   n  = nbase + nt * 16 + mlane;
        const float bv = bias[n];
#pragma unroll
        for (int r = 0; r < 8; ++r) {
            const int   row = mbase + r + moff;
            const float v   = acc[nt][r] + bv;
            size_t o;
            if (XPMODE) {
                // row = b*512 + t ; b = mt*16 + half*8 + rr
                const int t    = row & 511;
                const int b    = row >> 9;
                const int mt   = b >> 4;
                const int half = (b >> 3) & 1;
                const int rr   = b & 7;
                o = ((((size_t)t * 2 + mt) * 2 + half) * 256 + n) * 8 + rr;
            } else {
                o = (size_t)row * (size_t)N + n;
            }
            out[o] = v;
        }
    }
}

// ---------------------------------------------------------------------------
// Persistent single-workgroup RNN recurrence (the serial critical path):
//   h_t = tanh(xp[t] + h_{t-1} @ Wh),  h: [32 x 256], 512 steps
//   512 threads = 16 waves; wave w owns output columns [16w, 16w+16)
//   WhT B-operands live in registers for the whole scan; h double-buffered
//   in LDS; A chunks double-buffered so DS latency overlaps WMMA; xp is in
//   register layout (4x b128 loads / thread / step); split barrier overlaps
//   global h-stores with barrier arrival; fast exp2-based tanh.
// ---------------------------------------------------------------------------
__global__ __launch_bounds__(512) void k_rnn_scan(
    const float* __restrict__ xp,      // xp_s[(t*2+mt)*2+half][256 n][8 r] f32
    const __bf16* __restrict__ WhT,    // [256 n][256 k] bf16
    __bf16* __restrict__ hout,         // [b*T + t][256] bf16
    int T)
{
    __shared__ __bf16 hbuf[2][32 * 256];   // 32 KB double buffer

    const int tid  = threadIdx.x;
    const int lane = tid & 31;
    const int wave = tid >> 5;            // 0..15 -> ntile

    // h_{-1} = 0
    for (int i = tid; i < (32 * 256) / 2; i += 512)
        ((unsigned int*)hbuf[0])[i] = 0u;

    const int mlane = lane & 15;
    const int half  = lane >> 4;
    const int nloc  = wave * 16 + mlane;   // this lane's output column
    const int bhalf = half * 8;            // B K-offset per lane half
    const int ahalf = half * 8;            // A K-offset per lane half
    const int moff  = half * 8;            // C/D row offset per lane half

    // Preload Wh B-tiles for this wave's 16 columns: 8 K-chunks, kept in VGPRs
    V16 Bk[8];
#pragma unroll
    for (int kc = 0; kc < 8; ++kc) {
        const __bf16* p = WhT + (size_t)nloc * 256 + kc * 32 + bhalf;
        Bk[kc].q[0] = *(const uint4*)p;
        Bk[kc].q[1] = *(const uint4*)(p + 16);
    }
    __syncthreads();

    auto ldA = [&](int cur, int mt, int kc, V16& a) {
        const __bf16* p = &hbuf[cur][(mt * 16 + mlane) * 256 + kc * 32 + ahalf];
        a.q[0] = *(const uint4*)p;
        a.q[1] = *(const uint4*)(p + 16);
    };

    for (int t = 0; t < T; ++t) {
        const int cur = t & 1, nxt = cur ^ 1;

        // Stream this step's xp tile early (vectorized: 2x b128 per mt)
        F8 xv[2];
#pragma unroll
        for (int mt = 0; mt < 2; ++mt) {
            const float* p = xp +
                ((((size_t)t * 2 + mt) * 2 + half) * 256 + nloc) * 8;
            xv[mt].q[0] = *(const uint4*)p;
            xv[mt].q[1] = *(const uint4*)(p + 4);
        }

        v8f acc0 = v8f_zero();
        v8f acc1 = v8f_zero();

        // Double-buffered A chunks: DS loads for kc+1 fly under WMMAs for kc
        V16 a0[2], a1[2];
        ldA(cur, 0, 0, a0[0]);
        ldA(cur, 1, 0, a1[0]);
#pragma unroll
        for (int kc = 0; kc < 8; ++kc) {
            const int pb = kc & 1;
            if (kc < 7) {
                ldA(cur, 0, kc + 1, a0[pb ^ 1]);
                ldA(cur, 1, kc + 1, a1[pb ^ 1]);
            }
            acc0 = wmma_bf16(a0[pb], Bk[kc], acc0);
            acc1 = wmma_bf16(a1[pb], Bk[kc], acc1);
        }

        // h = tanh(xp + h@Wh); publish to next LDS buffer
        __bf16 hb[2][8];
#pragma unroll
        for (int mt = 0; mt < 2; ++mt) {
#pragma unroll
            for (int r = 0; r < 8; ++r) {
                const int   b  = mt * 16 + r + moff;
                const float av = (mt == 0) ? acc0[r] : acc1[r];
                hb[mt][r] = (__bf16)fast_tanh(av + xv[mt].f[r]);
                hbuf[nxt][b * 256 + nloc] = hb[mt][r];
            }
        }

#if __has_builtin(__builtin_amdgcn_s_barrier_signal) && \
    __has_builtin(__builtin_amdgcn_s_barrier_wait)
        // Split barrier: arrive once our LDS writes have landed, then do the
        // global stores while other waves finish arriving, then wait.
        asm volatile("s_wait_dscnt 0" ::: "memory");
        __builtin_amdgcn_s_barrier_signal(-1);
#pragma unroll
        for (int mt = 0; mt < 2; ++mt)
#pragma unroll
            for (int r = 0; r < 8; ++r) {
                const int b = mt * 16 + r + moff;
                hout[((size_t)b * T + t) * 256 + nloc] = hb[mt][r];
            }
        __builtin_amdgcn_s_barrier_wait(-1);
#else
#pragma unroll
        for (int mt = 0; mt < 2; ++mt)
#pragma unroll
            for (int r = 0; r < 8; ++r) {
                const int b = mt * 16 + r + moff;
                hout[((size_t)b * T + t) * 256 + nloc] = hb[mt][r];
            }
        __syncthreads();
#endif
    }
}

// ---------------------------------------------------------------------------
// Host-side orchestration
// ---------------------------------------------------------------------------
extern "C" void kernel_launch(void* const* d_in, const int* in_sizes, int n_in,
                              void* d_out, int out_size, void* d_ws, size_t ws_size,
                              hipStream_t stream) {
    (void)in_sizes; (void)n_in; (void)out_size; (void)ws_size;

    const int*   inp_ids = (const int*)  d_in[0];   // [32][512]
    const float* emb     = (const float*)d_in[1];   // [10000][128]
    const float* W1x     = (const float*)d_in[2];   // [128][256]
    const float* W1h     = (const float*)d_in[3];   // [256][256]
    const float* b1      = (const float*)d_in[4];   // [256]
    const float* W2x     = (const float*)d_in[5];   // [256][256]
    const float* W2h     = (const float*)d_in[6];   // [256][256]
    const float* b2      = (const float*)d_in[7];   // [256]
    const float* Wd      = (const float*)d_in[8];   // [256][10000]
    const float* bd      = (const float*)d_in[9];   // [10000]
    float* logits = (float*)d_out;                  // [32][512][10000]

    // Workspace carve-up (256B aligned)
    char* ws = (char*)d_ws;
    auto alloc = [&](size_t bytes) -> char* {
        char* p = ws;
        ws += (bytes + 255) & ~(size_t)255;
        return p;
    };
    __bf16* emb_bf = (__bf16*)alloc((size_t)10000 * 128 * 2);
    __bf16* W1xT   = (__bf16*)alloc((size_t)256 * 128 * 2);
    __bf16* W1hT   = (__bf16*)alloc((size_t)256 * 256 * 2);
    __bf16* W2xT   = (__bf16*)alloc((size_t)256 * 256 * 2);
    __bf16* W2hT   = (__bf16*)alloc((size_t)256 * 256 * 2);
    __bf16* WdT    = (__bf16*)alloc((size_t)10000 * 256 * 2);
    float*  xp     = (float*) alloc((size_t)16384 * 256 * 4);  // reused layer1/2
    __bf16* h1     = (__bf16*)alloc((size_t)16384 * 256 * 2);
    __bf16* h2     = (__bf16*)alloc((size_t)16384 * 256 * 2);

    const dim3 blk(256);

    // Precision staging: convert + transpose weights to bf16 B-layout
    k_convert_bf16<<<(10000 * 128 + 255) / 256, blk, 0, stream>>>(emb, emb_bf, 10000 * 128);
    k_transpose_bf16<<<(128 * 256  + 255) / 256, blk, 0, stream>>>(W1x, W1xT, 128, 256);
    k_transpose_bf16<<<(256 * 256  + 255) / 256, blk, 0, stream>>>(W1h, W1hT, 256, 256);
    k_transpose_bf16<<<(256 * 256  + 255) / 256, blk, 0, stream>>>(W2x, W2xT, 256, 256);
    k_transpose_bf16<<<(256 * 256  + 255) / 256, blk, 0, stream>>>(W2h, W2hT, 256, 256);
    k_transpose_bf16<<<(256 * 10000 + 255) / 256, blk, 0, stream>>>(Wd, WdT, 256, 10000);

    // xp1 = embed(ids) @ W1x + b1   (fused gather, M=16384, K=128, N=256)
    k_gemm<4, 128, true, true><<<dim3(128, 4), blk, (size_t)4 * 16 * 128 * 2, stream>>>(
        emb_bf, inp_ids, W1xT, b1, xp, 256);

    // layer-1 recurrence
    k_rnn_scan<<<1, 512, 0, stream>>>(xp, W1hT, h1, 512);

    // xp2 = h1 @ W2x + b2   (M=16384, K=256, N=256)
    k_gemm<4, 256, false, true><<<dim3(128, 4), blk, (size_t)4 * 16 * 256 * 2, stream>>>(
        h1, nullptr, W2xT, b2, xp, 256);

    // layer-2 recurrence
    k_rnn_scan<<<1, 512, 0, stream>>>(xp, W2hT, h2, 512);

    // logits = h2 @ Wd + bd   (M=16384, K=256, N=10000 = 125 * 80)
    k_gemm<5, 256, false, false><<<dim3(128, 125), blk, (size_t)5 * 16 * 256 * 2, stream>>>(
        h2, nullptr, WdT, bd, logits, 10000);
}
